// E3AttnBlk_21543555957272
// MI455X (gfx1250) — matrix operand order repaired and checked by reading, not compiled, not run on hardware
//
#include <hip/hip_runtime.h>
#include <hip/hip_bf16.h>
#include <math.h>

typedef __attribute__((ext_vector_type(16))) _Float16 v16h;
typedef __attribute__((ext_vector_type(8)))  float    v8f;

// ---- problem constants (match reference) ----
#define NB   2
#define NN   1024
#define KN   32
#define SPD  32
#define TDD  8
#define DDD  16
#define NH   4
#define NSd  4
#define NVd  2
#define NSCd 16
#define NVCd 8
#define NSId 40
#define MHD  64
#define WNUMd 1024
#define DIND 24

// output offsets (floats) in d_out, concatenated in return order
#define O1 0                         // out_s[:, :, :32]  (2*1024*32 = 65536)
#define O2 65536                     // out_v[:,:,0]      (2*1024*3  = 6144)
#define O3 71680                     // out_v[:,:,1]      (6144)
#define O4 77824                     // out_s[:, :, 32:]  (2*1024*8  = 16384)
#define O5 94208                     // attn              (2*4*1024*32 = 262144)

static __device__ __forceinline__ unsigned long long umin64(unsigned long long a,
                                                            unsigned long long b) {
  return a < b ? a : b;
}

// ------------------------------------------------------------------
// Kernel 1: kNN (K=32 smallest d2, stable tie-break on lower index,
// matching jax.lax.top_k(-d2, K) ordering). One block per (b,n).
// ------------------------------------------------------------------
__global__ void knn_kernel(const float* __restrict__ coord, int* __restrict__ idx) {
  __shared__ unsigned long long keys[NN];
  __shared__ unsigned long long red[256];
  const int node = blockIdx.x;
  const int b = node >> 10, n = node & 1023;
  const int tid = threadIdx.x;
  const float* cb = coord + (size_t)b * NN * 3;
  const float cx = cb[n * 3 + 0], cy = cb[n * 3 + 1], cz = cb[n * 3 + 2];
  for (int j = tid; j < NN; j += 256) {
    float dx = cb[j * 3 + 0] - cx, dy = cb[j * 3 + 1] - cy, dz = cb[j * 3 + 2] - cz;
    float d2 = dx * dx + dy * dy + dz * dz;                 // >= 0 -> bits monotonic
    keys[j] = (((unsigned long long)__float_as_uint(d2)) << 32) | (unsigned)j;
  }
  __syncthreads();
  for (int s = 0; s < KN; ++s) {
    unsigned long long m = ~0ULL;
    for (int j = tid; j < NN; j += 256) m = umin64(m, keys[j]);
    red[tid] = m;
    __syncthreads();
    for (int off = 128; off; off >>= 1) {
      if (tid < off) red[tid] = umin64(red[tid], red[tid + off]);
      __syncthreads();
    }
    if (tid == 0) {
      unsigned long long w = red[0];
      int jw = (int)(w & 0xffffffffu);
      idx[node * KN + s] = jw;
      keys[jw] = ~0ULL;
    }
    __syncthreads();
  }
}

// ------------------------------------------------------------------
// Kernel 2: convert Wk2/Wv2 [64][1024] f32 -> f16 WMMA-B fragments.
// Fragment order: out[((tile*2+frag)*32 + lane)*16 + e]
//   B (32x16, f16): N = lane&15, K = (lane>>4)*16 + e  (contiguous K order, same as the A fragments)
// ------------------------------------------------------------------
__global__ void swz_kernel(const float* __restrict__ Wk2, const float* __restrict__ Wv2,
                           _Float16* __restrict__ outK, _Float16* __restrict__ outV) {
  int g = blockIdx.x * 256 + threadIdx.x;
  if (g >= 64 * 2 * 32 * 16) return;
  int e = g & 15;
  int l = (g >> 4) & 31;
  int f = (g >> 9) & 1;
  int t = g >> 10;
  int K = ((l >> 4) << 4) + e;       // K within this 32-slice
  int m = f * 32 + K;                // hidden index 0..63
  int src = m * WNUMd + t * 16 + (l & 15);
  outK[g] = (_Float16)Wk2[src];
  outV[g] = (_Float16)Wv2[src];
}

// One MLP-2 output tile: D = A0*B0 + A1*B1 + bias  (bias exact via C fragment)
static __device__ __forceinline__ v8f wk_tile(const _Float16* __restrict__ Bsw,
                                              const float* __restrict__ b2,
                                              int tt, int l, int laneN,
                                              v16h a0, v16h a1) {
  v16h b0 = *(const v16h*)(Bsw + (size_t)((tt * 2 + 0) * 32 + l) * 16);
  v16h b1 = *(const v16h*)(Bsw + (size_t)((tt * 2 + 1) * 32 + l) * 16);
  float bias = b2[tt * 16 + laneN];
  v8f c;
#pragma unroll
  for (int r = 0; r < 8; ++r) c[r] = bias;
  c = __builtin_amdgcn_wmma_f32_16x16x32_f16(false, a0, false, b0, (short)0, c, false, false);
  c = __builtin_amdgcn_wmma_f32_16x16x32_f16(false, a1, false, b1, (short)0, c, false, false);
  return c;
}

// ------------------------------------------------------------------
// Kernel 3: fused per-node pipeline. One block (64 thr = 2 waves) per (b,n).
//   wave w owns edge M-tile w (edges 16w..16w+15) for the WMMA GEMM.
// ------------------------------------------------------------------
__global__ void __launch_bounds__(64)
fused_kernel(const float* __restrict__ sp, const float* __restrict__ coord,
             const float* __restrict__ veloc, const float* __restrict__ tG,
             const float* __restrict__ Wq_s, const float* __restrict__ Wq_v,
             const float* __restrict__ Wk1, const float* __restrict__ bk1,
             const float* __restrict__ bk2,
             const float* __restrict__ Wv1, const float* __restrict__ bv1,
             const float* __restrict__ bv2,
             const float* __restrict__ Wdot_s, const float* __restrict__ Wdot_v,
             const float* __restrict__ Wout_s, const float* __restrict__ Wout_v,
             const int* __restrict__ idxG,
             const _Float16* __restrict__ Bk, const _Float16* __restrict__ Bv,
             float* __restrict__ out) {
  // ---- LDS ----
  __shared__ float s_xs[NSId];             // xs_src: same for all 32 edges of node
  __shared__ float s_dotv[2][KN];
  __shared__ float s_xv[2][3][KN];
  __shared__ float s_cross[2][3][KN];
  __shared__ float s_shv[3][KN];
  __shared__ float s_dt[KN][DIND];
  __shared__ float s_W1[2][DIND][MHD];     // [k|v]
  __shared__ float s_b1[2][MHD];
  __shared__ __align__(32) _Float16 s_hA[2][2][2][32][16]; // [mlp][mtile][frag][lane][e]
  __shared__ float s_outS[2][KN][NSCd];    // ks / vs
  __shared__ float s_outV[2][KN][NVCd][3]; // kv / vv
  __shared__ float s_qs[KN][NSCd];
  __shared__ float s_qv[KN][NVCd][3];
  __shared__ float s_a[NH][KN];
  __shared__ float s_attn[NH][KN];
  __shared__ float s_osf[NSCd];
  __shared__ float s_ovf[NVCd][3];

  const int node = blockIdx.x;
  const int b = node >> 10, n = node & 1023;
  const int tid = threadIdx.x;
  const int* idxp = idxG + node * KN;

  // ---- Phase 0/1: preload W1/b1, per-edge geometry + d_t + qv, xs, qs ----
  for (int i = tid; i < DIND * MHD; i += 64) {
    (&s_W1[0][0][0])[i] = Wk1[i];
    (&s_W1[1][0][0])[i] = Wv1[i];
  }
  if (tid < MHD) { s_b1[0][tid] = bk1[tid]; s_b1[1][tid] = bv1[tid]; }

  const float* coordB = coord + (size_t)b * NN * 3;
  const float* velocB = veloc + (size_t)b * NN * 3;
  if (tid < KN) {
    const int e = tid;
    const int j = idxp[e];
    float cex = coordB[j * 3 + 0] - coordB[n * 3 + 0];
    float cey = coordB[j * 3 + 1] - coordB[n * 3 + 1];
    float cez = coordB[j * 3 + 2] - coordB[n * 3 + 2];
    float vex = velocB[j * 3 + 0] - velocB[n * 3 + 0];
    float vey = velocB[j * 3 + 1] - velocB[n * 3 + 1];
    float vez = velocB[j * 3 + 2] - velocB[n * 3 + 2];
    float d = sqrtf(cex * cex + cey * cey + cez * cez + 1e-12f);
    float rd = 1.f / d;
    float sx = cex * rd, sy = cey * rd, sz = cez * rd;      // shv (l=1 SH)
    s_shv[0][e] = sx; s_shv[1][e] = sy; s_shv[2][e] = sz;
    float x0x = -0.5f * cex, x0y = -0.5f * cey, x0z = -0.5f * cez;
    float x1x = -0.5f * vex, x1y = -0.5f * vey, x1z = -0.5f * vez;
    s_xv[0][0][e] = x0x; s_xv[0][1][e] = x0y; s_xv[0][2][e] = x0z;
    s_xv[1][0][e] = x1x; s_xv[1][1][e] = x1y; s_xv[1][2][e] = x1z;
    const float i3 = 0.57735027f, i2s = 0.70710678f;
    s_dotv[0][e] = (x0x * sx + x0y * sy + x0z * sz) * i3;
    s_dotv[1][e] = (x1x * sx + x1y * sy + x1z * sz) * i3;
    s_cross[0][0][e] = (x0y * sz - x0z * sy) * i2s;
    s_cross[0][1][e] = (x0z * sx - x0x * sz) * i2s;
    s_cross[0][2][e] = (x0x * sy - x0y * sx) * i2s;
    s_cross[1][0][e] = (x1y * sz - x1z * sy) * i2s;
    s_cross[1][1][e] = (x1z * sx - x1x * sz) * i2s;
    s_cross[1][2][e] = (x1x * sy - x1y * sx) * i2s;
#pragma unroll
    for (int m = 0; m < DDD; ++m) {        // gaussian soft one-hot, step=1/3
      float u = 3.f * d - (float)m;
      s_dt[e][m] = expf(-u * u) * (1.f / 1.12f);
    }
    const float* trowC = tG + ((size_t)b * NN + n) * TDD;
#pragma unroll
    for (int m = 0; m < TDD; ++m) s_dt[e][DDD + m] = trowC[m];
    // qv = (xv_dst[i] . Wq_v[i][o]) / sqrt(2), xv_dst = (ce/2, ve/2)
#pragma unroll
    for (int o = 0; o < NVCd; ++o) {
      float w0 = Wq_v[o], w1 = Wq_v[NVCd + o];
      s_qv[e][o][0] = (0.5f * cex * w0 + 0.5f * vex * w1) * i2s;
      s_qv[e][o][1] = (0.5f * cey * w0 + 0.5f * vey * w1) * i2s;
      s_qv[e][o][2] = (0.5f * cez * w0 + 0.5f * vez * w1) * i2s;
    }
  }
  if (tid < NSId) {
    s_xs[tid] = (tid < SPD) ? sp[((size_t)b * NN + n) * SPD + tid]
                            : tG[((size_t)b * NN + n) * TDD + (tid - SPD)];
  }
  // qs[e][o] = xs_dst . Wq_s / sqrt(40) ; thread group g covers o in [8g, 8g+8)
  {
    const int e = tid & 31, g = tid >> 5;
    const int j = idxp[e];
    const float* sprow = sp + ((size_t)b * NN + j) * SPD;
    const float* trow = tG + ((size_t)b * NN + j) * TDD;
    float acc[8] = {0, 0, 0, 0, 0, 0, 0, 0};
    for (int i = 0; i < NSId; ++i) {
      float x = (i < SPD) ? sprow[i] : trow[i - SPD];
#pragma unroll
      for (int oo = 0; oo < 8; ++oo) acc[oo] += x * Wq_s[i * NSCd + g * 8 + oo];
    }
#pragma unroll
    for (int oo = 0; oo < 8; ++oo) s_qs[e][g * 8 + oo] = acc[oo] * 0.15811388f;
  }
  __syncthreads();

  // ---- Phase 2: layer-1 MLP (f32 VALU) -> silu -> f16 A-fragments in LDS ----
  // A (16x32 f16): lane = (half<<4)|row, element e: K = half*16+e
  for (int task = tid; task < 2 * KN * MHD; task += 64) {
    int m = task & 63;
    int e = (task >> 6) & 31;
    int p = task >> 11;
    const float* W = &s_W1[p][0][0];
    float acc = s_b1[p][m];
#pragma unroll
    for (int i = 0; i < DIND; ++i) acc += s_dt[e][i] * W[i * MHD + m];
    float h = acc / (1.f + expf(-acc));    // silu
    int mt = e >> 4, row = e & 15, f = m >> 5, k = m & 31;
    int half = k >> 4;
    int el = k & 15;
    s_hA[p][mt][f][half * 16 + row][el] = (_Float16)h;
  }
  __syncthreads();

  // ---- Phase 3: WMMA GEMM (h[16x64] x W2-tile[64x16]) + fused TP epilogue ----
  // Region-split loops: straight-line bodies so the scheduler can pipeline
  // next tile's global_load_b128 B-fragments under the current WMMAs.
  {
    const int l = tid & 31;
    const int mt = tid >> 5;               // this wave's edge M-tile
    const int laneN = l & 15;
    const int half = l >> 4;
    const int ebase = mt * 16 + half * 8;  // first of the 8 edges this lane's D rows cover
    for (int p = 0; p < 2; ++p) {
      const _Float16* Bsw = p ? Bv : Bk;
      const float* b2 = p ? bv2 : bk2;
      v16h a0 = *(const v16h*)&s_hA[p][mt][0][l][0];
      v16h a1 = *(const v16h*)&s_hA[p][mt][1][l][0];
      float accS[8], accSV[8], accV[8][3];
#pragma unroll
      for (int r = 0; r < 8; ++r) {
        accS[r] = 0.f; accSV[r] = 0.f;
        accV[r][0] = 0.f; accV[r][1] = 0.f; accV[r][2] = 0.f;
      }
      // --- region ss: tiles 0..39, cols i*16+o, coeff xs[i] (lane-uniform) ---
#pragma unroll 2
      for (int tt = 0; tt < 40; ++tt) {
        v8f c = wk_tile(Bsw, b2, tt, l, laneN, a0, a1);
        float cf = s_xs[tt];
#pragma unroll
        for (int r = 0; r < 8; ++r) accS[r] += cf * c[r];
      }
      // --- region vv0: tiles 40..41, coeff dotv[i][edge] (per-row) ---
#pragma unroll
      for (int tt = 40; tt < 42; ++tt) {
        v8f c = wk_tile(Bsw, b2, tt, l, laneN, a0, a1);
        const float* dv = &s_dotv[tt - 40][ebase];
#pragma unroll
        for (int r = 0; r < 8; ++r) accS[r] += dv[r] * c[r];
      }
      // --- region sv: tiles 42..61, coeff xs[(tt-42)*2 + laneN>>3] ---
#pragma unroll 2
      for (int tt = 42; tt < 62; ++tt) {
        v8f c = wk_tile(Bsw, b2, tt, l, laneN, a0, a1);
        float cf = s_xs[(tt - 42) * 2 + (laneN >> 3)];
#pragma unroll
        for (int r = 0; r < 8; ++r) accSV[r] += cf * c[r];
      }
      // --- tile 62: w_v0, coeff xv[i][c][edge] ---
      {
        v8f c = wk_tile(Bsw, b2, 62, l, laneN, a0, a1);
        const int i = laneN >> 3;
#pragma unroll
        for (int cc = 0; cc < 3; ++cc) {
          const float* xv = &s_xv[i][cc][ebase];
#pragma unroll
          for (int r = 0; r < 8; ++r) accV[r][cc] += xv[r] * c[r];
        }
      }
      // --- tile 63: w_vxv, coeff crossv[i][c][edge] ---
      {
        v8f c = wk_tile(Bsw, b2, 63, l, laneN, a0, a1);
        const int i = laneN >> 3;
#pragma unroll
        for (int cc = 0; cc < 3; ++cc) {
          const float* cv = &s_cross[i][cc][ebase];
#pragma unroll
          for (int r = 0; r < 8; ++r) accV[r][cc] += cv[r] * c[r];
        }
      }
      const float i42 = 0.15430335f;       // 1/sqrt(42)
#pragma unroll
      for (int r = 0; r < 8; ++r) s_outS[p][ebase + r][laneN] = accS[r] * i42;
      // fold lane pairs (l, l^8): two i-contributions per (edge, o)
#pragma unroll
      for (int r = 0; r < 8; ++r) accSV[r] += __shfl_xor(accSV[r], 8);
#pragma unroll
      for (int r = 0; r < 8; ++r)
#pragma unroll
        for (int cc = 0; cc < 3; ++cc) accV[r][cc] += __shfl_xor(accV[r][cc], 8);
      if ((l & 8) == 0) {
        const int o = laneN & 7;
        const float i44 = 0.15075567f;     // 1/sqrt(44)
#pragma unroll
        for (int r = 0; r < 8; ++r) {
          int e2 = ebase + r;
#pragma unroll
          for (int cc = 0; cc < 3; ++cc)
            s_outV[p][e2][o][cc] = (accSV[r] * s_shv[cc][e2] + accV[r][cc]) * i44;
        }
      }
    }
  }
  __syncthreads();

  // ---- Phase 4: attention logits ----
  for (int task = tid; task < NH * KN; task += 64) {
    int h = task >> 5, e = task & 31;
    float acc = 0.f;
#pragma unroll
    for (int s = 0; s < NSd; ++s)
#pragma unroll
      for (int t4 = 0; t4 < NSd; ++t4)
        acc += s_qs[e][h * NSd + s] * Wdot_s[s * NSd + t4] * s_outS[0][e][h * NSd + t4];
    float accv = 0.f;
#pragma unroll
    for (int i = 0; i < NVd; ++i)
#pragma unroll
      for (int j2 = 0; j2 < NVd; ++j2) {
        float w = Wdot_v[i * NVd + j2];
#pragma unroll
        for (int cc = 0; cc < 3; ++cc)
          accv += s_qv[e][h * NVd + i][cc] * w * s_outV[0][e][h * NVd + j2][cc];
      }
    acc += accv * 0.57735027f;             // /sqrt(3)
    s_a[h][e] = acc * 0.070710678f;        // /sqrt(20)/sqrt(10)
  }
  __syncthreads();

  // ---- Phase 5: softmax over K per head; emit attn output ----
  if (tid < NH) {
    const int h = tid;
    float m = -3.4e38f;
    for (int e = 0; e < KN; ++e) m = fmaxf(m, s_a[h][e]);
    float ssum = 0.f;
    for (int e = 0; e < KN; ++e) {
      float x = expf(s_a[h][e] - m);
      s_attn[h][e] = x;
      ssum += x;
    }
    float inv = 1.f / ssum;
    float* dst = out + O5 + (((size_t)b * NH + h) * NN + n) * KN;
    for (int e = 0; e < KN; ++e) {
      float w = s_attn[h][e] * inv;
      s_attn[h][e] = w;
      dst[e] = w;
    }
  }
  __syncthreads();

  // ---- Phase 6: attention-weighted sums of values ----
  if (tid < NSCd) {
    const int o = tid;
    float acc = 0.f;
    for (int e = 0; e < KN; ++e) acc += s_attn[o >> 2][e] * s_outS[1][e][o];
    s_osf[o] = acc;
  } else if (tid < NSCd + NVCd * 3) {
    const int q = tid - NSCd;
    const int o = q / 3, cc = q % 3;
    float acc = 0.f;
    for (int e = 0; e < KN; ++e) acc += s_attn[o >> 1][e] * s_outV[1][e][o][cc];
    s_ovf[o][cc] = acc;
  }
  __syncthreads();

  // ---- Phase 7: output projections ----
  if (tid < NSId) {
    float acc = 0.f;
#pragma unroll
    for (int o = 0; o < NSCd; ++o) acc += s_osf[o] * Wout_s[o * NSId + tid];
    acc *= 0.25f;                          // 1/sqrt(16)
    if (tid < SPD) out[O1 + ((size_t)b * NN + n) * SPD + tid] = acc;
    else out[O4 + ((size_t)b * NN + n) * TDD + (tid - SPD)] = acc;
  } else if (tid < NSId + 6) {
    const int q = tid - NSId;
    const int oo = q / 3, cc = q % 3;
    float acc = 0.f;
#pragma unroll
    for (int o = 0; o < NVCd; ++o) acc += s_ovf[o][cc] * Wout_v[o * 2 + oo];
    acc *= 0.35355339f;                    // 1/sqrt(8)
    out[(oo == 0 ? O2 : O3) + ((size_t)b * NN + n) * 3 + cc] = acc;
  }
}

// ------------------------------------------------------------------
extern "C" void kernel_launch(void* const* d_in, const int* in_sizes, int n_in,
                              void* d_out, int out_size, void* d_ws, size_t ws_size,
                              hipStream_t stream) {
  (void)in_sizes; (void)n_in; (void)out_size; (void)ws_size;
  const float* sp     = (const float*)d_in[0];
  const float* coord  = (const float*)d_in[1];
  const float* veloc  = (const float*)d_in[2];
  const float* tG     = (const float*)d_in[3];
  const float* Wq_s   = (const float*)d_in[4];
  const float* Wq_v   = (const float*)d_in[5];
  const float* Wk1    = (const float*)d_in[6];
  const float* bk1    = (const float*)d_in[7];
  const float* Wk2    = (const float*)d_in[8];
  const float* bk2    = (const float*)d_in[9];
  const float* Wv1    = (const float*)d_in[10];
  const float* bv1    = (const float*)d_in[11];
  const float* Wv2    = (const float*)d_in[12];
  const float* bv2    = (const float*)d_in[13];
  const float* Wdot_s = (const float*)d_in[14];
  const float* Wdot_v = (const float*)d_in[15];
  const float* Wout_s = (const float*)d_in[16];
  const float* Wout_v = (const float*)d_in[17];

  char* ws = (char*)d_ws;
  int* idx       = (int*)ws;                               // 2048*32*4   = 256 KB
  _Float16* Bk   = (_Float16*)(ws + 262144);               // 64K f16     = 128 KB
  _Float16* Bv   = (_Float16*)(ws + 262144 + 131072);      //             = 128 KB
  float* outp    = (float*)d_out;

  knn_kernel<<<NB * NN, 256, 0, stream>>>(coord, idx);
  swz_kernel<<<(64 * 2 * 32 * 16 + 255) / 256, 256, 0, stream>>>(Wk2, Wv2, Bk, Bv);
  fused_kernel<<<NB * NN, 64, 0, stream>>>(sp, coord, veloc, tG, Wq_s, Wq_v,
                                           Wk1, bk1, bk2, Wv1, bv1, bv2,
                                           Wdot_s, Wdot_v, Wout_s, Wout_v,
                                           idx, Bk, Bv, outp);
}
